// SetAttention_Linear_18605798326480
// MI455X (gfx1250) — matrix operand order repaired and checked by reading, not compiled, not run on hardware
//
#include <hip/hip_runtime.h>

#define Bn 2
#define Tn 2048
#define Cn 1024
#define NHn 16
#define HSn 64
#define LVL 3
#define STOT 511   // sum_{l=3..11} 2048>>l

#define BM 128     // block tile M (8 waves x 16 rows)
#define BN 64      // block tile N
#define BK 32      // k-slab per WMMA step

typedef __attribute__((ext_vector_type(16))) __bf16 v16bf;
typedef __attribute__((ext_vector_type(8)))  float  v8f;
typedef __attribute__((ext_vector_type(4)))  int    v4i;
typedef __attribute__((ext_vector_type(8)))  int    v8i;

__device__ __forceinline__ unsigned short f2bf(float f) {
  union { float f; unsigned u; } v; v.f = f;
  unsigned r = v.u + 0x7fffu + ((v.u >> 16) & 1u);   // round-to-nearest-even
  return (unsigned short)(r >> 16);
}

union FragU { v16bf v; uint4 u[2]; };

// A-matrix 16x32 bf16 fragment (row M = lane%16):
// lanes 0-15 hold K = [0..15]; lanes 16-31 hold K = [16..31].
__device__ __forceinline__ v16bf load_fragA(const unsigned short* row, int hi) {
  FragU f;
  int o = hi ? 16 : 0;
  f.u[0] = *reinterpret_cast<const uint4*>(row + o);
  f.u[1] = *reinterpret_cast<const uint4*>(row + o + 8);
  return f.v;
}
// B-matrix 32x16 bf16 fragment (col N = lane%16):
// lanes 0-15 hold K = [0..15]; lanes 16-31 hold K = [16..31].
__device__ __forceinline__ v16bf load_fragB(const unsigned short* row, int hi) {
  FragU f;
  int o = hi ? 16 : 0;
  f.u[0] = *reinterpret_cast<const uint4*>(row + o);
  f.u[1] = *reinterpret_cast<const uint4*>(row + o + 8);
  return f.v;
}

// Tensor Data Mover: DMA a 2D tile (BK elements wide x `tile_rows` rows,
// row stride = K elements, bf16) from global into LDS at `lds_off`.
// D# group0 (4 SGPRs) + group1 (8 SGPRs), 2D form (VADDR2/3 omitted).
__device__ __forceinline__ void tdm_load_tile(unsigned lds_off,
                                              const unsigned short* gptr,
                                              int tensor_rows, int tile_rows,
                                              int krow) {
  unsigned long long ga = (unsigned long long)(size_t)gptr;
  v4i g0;
  g0[0] = 1;                                           // count=1 (valid user D#)
  g0[1] = (int)lds_off;                                // lds_addr (bytes)
  g0[2] = (int)(unsigned)(ga & 0xffffffffull);         // global_addr[31:0]
  g0[3] = (int)(((ga >> 32) & 0x1ffffffull) | (2u << 30)); // [56:32] | type=2
  v8i g1;
  g1[0] = 1 << 16;                                     // wg_mask=0, data_size=1 (2B)
  g1[1] = (krow & 0xffff) << 16;                       // tensor_dim0[15:0]
  g1[2] = ((krow >> 16) & 0xffff) | ((tensor_rows & 0xffff) << 16); // dim0 hi | dim1 lo
  g1[3] = ((tensor_rows >> 16) & 0xffff) | (BK << 16); // dim1 hi | tile_dim0
  g1[4] = tile_rows;                                   // tile_dim1 | tile_dim2=0
  g1[5] = krow;                                        // tensor_dim0_stride[31:0]
  g1[6] = 0;
  g1[7] = 0;
  asm volatile("tensor_load_to_lds %0, %1" :: "s"(g0), "s"(g1) : "memory");
}

__global__ void f32_to_bf16(const float* __restrict__ s, unsigned short* __restrict__ d, int n) {
  int i = blockIdx.x * 256 + threadIdx.x;
  if (i < n) d[i] = f2bf(s[i]);
}

// Shared 8-wave 128x64 WMMA tile engine: TDM double-buffered LDS k-slabs.
// Returns 4 accumulators (16x16 each) for this wave's 16(M)x64(N) strip.
__device__ __forceinline__ void wmma_tile_128x64(
    const unsigned short* __restrict__ A, int Mtot,
    const unsigned short* __restrict__ Bw, int Ntot,
    int mt, int nt, int wv, int sub, int hi,
    unsigned short (*smA)[BM * BK], unsigned short (*smB)[BN * BK],
    v8f& c0, v8f& c1, v8f& c2, v8f& c3) {
  const int K = Cn;
  const int nsteps = K / BK;                  // 32
  bool issuer = (wv == 0);
  if (issuer) {
    tdm_load_tile((unsigned)(size_t)(void*)smA[0], A + (size_t)mt * K, Mtot, BM, K);
    tdm_load_tile((unsigned)(size_t)(void*)smB[0], Bw + (size_t)nt * K, Ntot, BN, K);
  }
  for (int ks = 0; ks < nsteps; ++ks) {
    int cur = ks & 1;
    if (issuer) {
      if (ks + 1 < nsteps) {
        int nxt = cur ^ 1;
        tdm_load_tile((unsigned)(size_t)(void*)smA[nxt],
                      A + (size_t)mt * K + (ks + 1) * BK, Mtot, BM, K);
        tdm_load_tile((unsigned)(size_t)(void*)smB[nxt],
                      Bw + (size_t)nt * K + (ks + 1) * BK, Ntot, BN, K);
        __builtin_amdgcn_s_wait_tensorcnt(2);  // current slab done; next 2 in flight
      } else {
        __builtin_amdgcn_s_wait_tensorcnt(0);  // final slab done
      }
    }
    __syncthreads();
    const unsigned short* ta = &smA[cur][((wv << 4) + sub) * BK];
    const unsigned short* tb = &smB[cur][sub * BK];
    v16bf a  = load_fragA(ta, hi);
    v16bf b0 = load_fragB(tb,            hi);
    v16bf b1 = load_fragB(tb + 16 * BK,  hi);
    v16bf b2 = load_fragB(tb + 32 * BK,  hi);
    v16bf b3 = load_fragB(tb + 48 * BK,  hi);
    c0 = __builtin_amdgcn_wmma_f32_16x16x32_bf16(false, a, false, b0, (short)0, c0, false, false);
    c1 = __builtin_amdgcn_wmma_f32_16x16x32_bf16(false, a, false, b1, (short)0, c1, false, false);
    c2 = __builtin_amdgcn_wmma_f32_16x16x32_bf16(false, a, false, b2, (short)0, c2, false, false);
    c3 = __builtin_amdgcn_wmma_f32_16x16x32_bf16(false, a, false, b3, (short)0, c3, false, false);
    __syncthreads();
  }
}

// qkv = x @ W_attn^T ; fused epilogue: scatter q / q_phi / k_phi / v per-head.
__global__ __launch_bounds__(256) void gemm_qkv_wmma(
    const unsigned short* __restrict__ A, const unsigned short* __restrict__ Bw,
    float* __restrict__ qg, float* __restrict__ qph,
    float* __restrict__ kc, float* __restrict__ vc) {
  __shared__ unsigned short smA[2][BM * BK];   // 2 x 8KB
  __shared__ unsigned short smB[2][BN * BK];   // 2 x 4KB
  int mt = blockIdx.y * BM;
  int nt = blockIdx.x * BN;
  int tid = threadIdx.x;
  int wv = tid >> 5, lane = tid & 31;
  int sub = lane & 15, hi = lane >> 4;
  v8f c0 = {}, c1 = {}, c2 = {}, c3 = {};
  wmma_tile_128x64(A, Bn * Tn, Bw, 3 * Cn, mt, nt, wv, sub, hi, smA, smB, c0, c1, c2, c3);
  v8f accs[4] = {c0, c1, c2, c3};
#pragma unroll
  for (int j = 0; j < 4; ++j) {
#pragma unroll
    for (int i = 0; i < 8; ++i) {
      float val = accs[j][i];
      int m = mt + (wv << 4) + i + (hi << 3);  // C/D layout: VGPR i -> M=i (+8 hi half)
      int n = nt + (j << 4) + sub;             // N = lane%16
      int b = m >> 11, t = m & (Tn - 1);
      int sec = n >> 10, ci = n & (Cn - 1);
      int h = ci >> 6, d = ci & 63;
      size_t base = ((((size_t)b * NHn + h) * Tn + t) << 6) + d;
      if (sec == 0) {
        qg[base]  = val;
        qph[base] = val > 0.f ? val + 1.f : __expf(val);   // elu(x)+1
      } else if (sec == 1) {
        kc[base] = val > 0.f ? val + 1.f : __expf(val);    // phi(k), pre-cumsum
      } else {
        vc[base] = val;
      }
    }
  }
}

// out = ctx @ W_proj^T (f32 output)
__global__ __launch_bounds__(256) void gemm_proj_wmma(
    const unsigned short* __restrict__ A, const unsigned short* __restrict__ Bw,
    float* __restrict__ out) {
  __shared__ unsigned short smA[2][BM * BK];
  __shared__ unsigned short smB[2][BN * BK];
  int mt = blockIdx.y * BM;
  int nt = blockIdx.x * BN;
  int tid = threadIdx.x;
  int wv = tid >> 5, lane = tid & 31;
  int sub = lane & 15, hi = lane >> 4;
  v8f c0 = {}, c1 = {}, c2 = {}, c3 = {};
  wmma_tile_128x64(A, Bn * Tn, Bw, Cn, mt, nt, wv, sub, hi, smA, smB, c0, c1, c2, c3);
  v8f accs[4] = {c0, c1, c2, c3};
#pragma unroll
  for (int j = 0; j < 4; ++j) {
#pragma unroll
    for (int i = 0; i < 8; ++i) {
      int m = mt + (wv << 4) + i + (hi << 3);
      int n = nt + (j << 4) + sub;
      out[(size_t)m * Cn + n] = accs[j][i];
    }
  }
}

// In-place cumsum over T for k_phi and v; one block per (b,h), thread per dim.
__global__ void cumsum_k(float* __restrict__ kc, float* __restrict__ vc) {
  int bh = blockIdx.x;
  int d = threadIdx.x;
  float* kp = kc + (((size_t)bh * Tn) << 6) + d;
  float* vp = vc + (((size_t)bh * Tn) << 6) + d;
  float ak = 0.f, av = 0.f;
  for (int t = 0; t < Tn; ++t) {
    size_t o = (size_t)t << 6;
    ak += kp[o]; kp[o] = ak;
    av += vp[o]; vp[o] = av;
  }
}

// One workgroup per (b,h). K_set (511x64 f32 = 131KB) staged in LDS; V_set
// reconstructed from L2-resident v_cum. Half-wave (16 lanes x 4 dims) per t,
// online softmax over valid sets (contiguous per level) + tail segment.
__global__ __launch_bounds__(256) void attn_kernel(
    const float* __restrict__ qg, const float* __restrict__ qph,
    const float* __restrict__ kc, const float* __restrict__ vc,
    unsigned short* __restrict__ ctx) {
  __shared__ float Ks[STOT * HSn];
  int bh = blockIdx.x;
  const float* kcb = kc  + (((size_t)bh * Tn) << 6);
  const float* vcb = vc  + (((size_t)bh * Tn) << 6);
  const float* qb  = qg  + (((size_t)bh * Tn) << 6);
  const float* qpb = qph + (((size_t)bh * Tn) << 6);

  for (int idx = threadIdx.x; idx < STOT * HSn; idx += 256) {
    int s = idx >> 6, d = idx & 63;
    int rem = s, l = LVL;
    while (rem >= (Tn >> l)) { rem -= (Tn >> l); ++l; }
    int li = rem << l;
    int ri = li + (1 << l) - 1;
    float kv = kcb[((size_t)ri << 6) + d];
    if (li >= 1) kv -= kcb[((size_t)(li - 1) << 6) + d];
    Ks[idx] = kv;
  }
  __syncthreads();

  int off[13];
  off[LVL] = 0;
  for (int l = LVL; l <= 11; ++l) off[l + 1] = off[l] + (Tn >> l);

  int w = threadIdx.x >> 5, lane = threadIdx.x & 31;
  int half = lane >> 4, sub = lane & 15;
  const float scl = 0.125f;   // hs^-0.5
  int b = bh >> 4, h = bh & 15;

  for (int t = (w << 1) + half; t < Tn; t += 16) {
    int qo = (t << 6) + (sub << 2);
    float4 qv = *(const float4*)(qb + qo);
    float4 qp = *(const float4*)(qpb + qo);
    // tail segment: [t - t%8, t]
    int lt1 = t - (t & 7) - 1;
    float4 kt = *(const float4*)(kcb + qo);
    float4 vt = *(const float4*)(vcb + qo);
    if (lt1 >= 0) {
      float4 k2 = *(const float4*)(kcb + (((size_t)lt1) << 6) + (sub << 2));
      float4 v2 = *(const float4*)(vcb + (((size_t)lt1) << 6) + (sub << 2));
      kt.x -= k2.x; kt.y -= k2.y; kt.z -= k2.z; kt.w -= k2.w;
      vt.x -= v2.x; vt.y -= v2.y; vt.z -= v2.z; vt.w -= v2.w;
    }
    float part = qp.x*kt.x + qp.y*kt.y + qp.z*kt.z + qp.w*kt.w;
#pragma unroll
    for (int o = 1; o < 16; o <<= 1) part += __shfl_xor(part, o, 16);
    float mrun = part * scl;                 // tail logit is softmax baseline
    float ssum = 1.f;
    float ax = vt.x, ay = vt.y, az = vt.z, aw = vt.w;

    for (int l = LVL; l <= 11; ++l) {
      int cnt = (t + 1) >> l;                // sets with r <= t are a prefix
      int sbase = off[l];
      for (int j = 0; j < cnt; ++j) {
        int s = sbase + j;
        float4 ksv = *(const float4*)(&Ks[(s << 6) + (sub << 2)]);
        float pl = qv.x*ksv.x + qv.y*ksv.y + qv.z*ksv.z + qv.w*ksv.w;
#pragma unroll
        for (int o = 1; o < 16; o <<= 1) pl += __shfl_xor(pl, o, 16);
        float logit = pl * scl;
        float nm = fmaxf(mrun, logit);
        float eo = __expf(mrun - nm);
        float p  = __expf(logit - nm);
        int li = j << l;
        int ri = li + (1 << l) - 1;
        float4 vsv = *(const float4*)(vcb + (((size_t)ri) << 6) + (sub << 2));
        if (li >= 1) {
          float4 v2 = *(const float4*)(vcb + (((size_t)(li - 1)) << 6) + (sub << 2));
          vsv.x -= v2.x; vsv.y -= v2.y; vsv.z -= v2.z; vsv.w -= v2.w;
        }
        ssum = ssum * eo + p;
        ax = ax * eo + p * vsv.x;
        ay = ay * eo + p * vsv.y;
        az = az * eo + p * vsv.z;
        aw = aw * eo + p * vsv.w;
        mrun = nm;
      }
    }
    float inv = 1.f / ssum;
    size_t ob = (((size_t)b * Tn + t) << 10) + (h << 6) + (sub << 2);
    ushort4 o4;
    o4.x = f2bf(ax * inv); o4.y = f2bf(ay * inv);
    o4.z = f2bf(az * inv); o4.w = f2bf(aw * inv);
    *reinterpret_cast<ushort4*>(ctx + ob) = o4;
  }
}

extern "C" void kernel_launch(void* const* d_in, const int* in_sizes, int n_in,
                              void* d_out, int out_size, void* d_ws, size_t ws_size,
                              hipStream_t stream) {
  const float* x  = (const float*)d_in[0];
  const float* Wa = (const float*)d_in[1];
  const float* Wp = (const float*)d_in[2];
  float* out = (float*)d_out;

  char* ws = (char*)d_ws;
  size_t o = 0;
  unsigned short* xb  = (unsigned short*)(ws + o); o += (size_t)Bn * Tn * Cn * 2;
  unsigned short* Wab = (unsigned short*)(ws + o); o += (size_t)3 * Cn * Cn * 2;
  unsigned short* Wpb = (unsigned short*)(ws + o); o += (size_t)Cn * Cn * 2;
  float* qg  = (float*)(ws + o); o += (size_t)Bn * Tn * Cn * 4;
  float* qph = (float*)(ws + o); o += (size_t)Bn * Tn * Cn * 4;
  float* kc  = (float*)(ws + o); o += (size_t)Bn * Tn * Cn * 4;
  float* vc  = (float*)(ws + o); o += (size_t)Bn * Tn * Cn * 4;
  unsigned short* ctx = (unsigned short*)(ws + o); o += (size_t)Bn * Tn * Cn * 2;
  (void)ws_size; (void)in_sizes; (void)n_in; (void)out_size;

  int n1 = Bn * Tn * Cn;
  f32_to_bf16<<<(n1 + 255) / 256, 256, 0, stream>>>(x, xb, n1);
  int n2 = 3 * Cn * Cn;
  f32_to_bf16<<<(n2 + 255) / 256, 256, 0, stream>>>(Wa, Wab, n2);
  int n3 = Cn * Cn;
  f32_to_bf16<<<(n3 + 255) / 256, 256, 0, stream>>>(Wp, Wpb, n3);

  dim3 gq(3 * Cn / BN, Bn * Tn / BM);          // (48, 32)
  gemm_qkv_wmma<<<gq, 256, 0, stream>>>(xb, Wab, qg, qph, kc, vc);

  cumsum_k<<<Bn * NHn, HSn, 0, stream>>>(kc, vc);

  attn_kernel<<<Bn * NHn, 256, 0, stream>>>(qg, qph, kc, vc, ctx);

  dim3 gp(Cn / BN, Bn * Tn / BM);              // (16, 32)
  gemm_proj_wmma<<<gp, 256, 0, stream>>>(ctx, Wpb, out);
}
